// RingAttentionHelper_38242388804062
// MI455X (gfx1250) — compile-verified, matrix operand order checked by reading, not run on hardware
//
#include <hip/hip_runtime.h>

// ---------------------------------------------------------------------------
// Fused LayerNorm + QKV projection + RoPE + GQA-expand for MI455X (gfx1250).
//
// bf16x2-split WMMA GEMM (hi*hi + hi*lo + lo*hi) for ~f32 accuracy at the
// bf16 matrix-engine rate. Double-buffered LDS staging via CDNA5
// GLOBAL_LOAD_ASYNC_TO_LDS_B128 (ASYNCcnt-tracked) when the builtin is
// available; register-staging fallback otherwise. Epilogue applies
// rotate-half RoPE and stores Q / K(x4 GQA) / V(x4) in [B,h,T,DH] layout.
// ---------------------------------------------------------------------------

typedef __bf16 bf16_t;
typedef __attribute__((ext_vector_type(16))) __bf16 v16bf;
typedef __attribute__((ext_vector_type(8)))  __bf16 v8bf;
typedef __attribute__((ext_vector_type(8)))  float  v8f;
typedef __attribute__((ext_vector_type(4)))  int    v4i;

constexpr int kB   = 2;
constexpr int kT   = 4096;
constexpr int kD   = 4096;           // model dim == GEMM K
constexpr int kH   = 32;
constexpr int kKVH = 8;
constexpr int kDH  = 128;
constexpr int kNQ  = kH * kDH;       // 4096
constexpr int kNK  = kKVH * kDH;     // 1024
constexpr int kNT  = kNQ + 2 * kNK;  // 6144 fused N
constexpr int kM   = kB * kT;        // 8192 rows

#if defined(__gfx1250__) && __has_builtin(__builtin_amdgcn_global_load_async_to_lds_b128)
#define USE_ASYNC_LDS 1
#else
#define USE_ASYNC_LDS 0
#endif

#if USE_ASYNC_LDS
// Builtin signature (from hipcc diagnostic): (v4i AS1*, v4i AS3*, imm, imm)
#define GASYNC_B128(g, l)                                                    \
  __builtin_amdgcn_global_load_async_to_lds_b128(                            \
      (__attribute__((address_space(1))) v4i*)(g),                           \
      (__attribute__((address_space(3))) v4i*)(l), 0, 0)
#if __has_builtin(__builtin_amdgcn_s_wait_asynccnt)
#define ASYNC_WAIT(n) __builtin_amdgcn_s_wait_asynccnt(n)
#else
#define ASYNC_WAIT(n) asm volatile("s_wait_asynccnt %0" ::"n"(n) : "memory")
#endif
#endif

// ---------------------------------------------------------------------------
// Kernel 1: LayerNorm row -> bf16 hi/lo split
// ---------------------------------------------------------------------------
__global__ __launch_bounds__(256)
void ln_to_bf16(const float* __restrict__ x,
                const float* __restrict__ gamma,
                const float* __restrict__ beta,
                bf16_t* __restrict__ xhi, bf16_t* __restrict__ xlo) {
  const int row = blockIdx.x;
  const int tid = threadIdx.x;                 // 256 threads, 16 cols each
  const float* xr = x + (size_t)row * kD;

  float v[16];
  float s = 0.f, s2 = 0.f;
#pragma unroll
  for (int i = 0; i < 16; ++i) {
    float f = xr[tid + 256 * i];
    v[i] = f; s += f; s2 += f * f;
  }
  __shared__ float red0[256], red1[256];
  red0[tid] = s; red1[tid] = s2;
  __syncthreads();
  for (int off = 128; off > 0; off >>= 1) {
    if (tid < off) { red0[tid] += red0[tid + off]; red1[tid] += red1[tid + off]; }
    __syncthreads();
  }
  const float mu   = red0[0] * (1.0f / kD);
  const float var  = red1[0] * (1.0f / kD) - mu * mu;
  const float rstd = rsqrtf(var + 1e-5f);

  bf16_t* oh = xhi + (size_t)row * kD;
  bf16_t* ol = xlo + (size_t)row * kD;
#pragma unroll
  for (int i = 0; i < 16; ++i) {
    const int c = tid + 256 * i;
    const float y = (v[i] - mu) * rstd * gamma[c] + beta[c];
    const bf16_t h = (bf16_t)y;
    const bf16_t l = (bf16_t)(y - (float)h);
    oh[c] = h; ol[c] = l;
  }
}

// ---------------------------------------------------------------------------
// Kernel 2: W [K=4096][N] f32  ->  WT [N][K] bf16 hi/lo (tiled transpose)
// ---------------------------------------------------------------------------
__global__ __launch_bounds__(256)
void wT_to_bf16(const float* __restrict__ W, int N,
                bf16_t* __restrict__ whi, bf16_t* __restrict__ wlo) {
  __shared__ float tile[32][33];
  const int n0 = blockIdx.x * 32;   // along N
  const int k0 = blockIdx.y * 32;   // along K (=kD)
  const int tx = threadIdx.x & 31;
  const int ty = threadIdx.x >> 5;  // 32x8
#pragma unroll
  for (int i = 0; i < 4; ++i) {
    const int k = k0 + ty + 8 * i;
    tile[ty + 8 * i][tx] = W[(size_t)k * N + (n0 + tx)];
  }
  __syncthreads();
#pragma unroll
  for (int i = 0; i < 4; ++i) {
    const int n = n0 + ty + 8 * i;
    const float f = tile[tx][ty + 8 * i];
    const bf16_t h = (bf16_t)f;
    const bf16_t l = (bf16_t)(f - (float)h);
    whi[(size_t)n * kD + (k0 + tx)] = h;
    wlo[(size_t)n * kD + (k0 + tx)] = l;
  }
}

// ---------------------------------------------------------------------------
// Kernel 3: WMMA GEMM (128x128x32 tiles, bf16x2 split) + RoPE + GQA store
// ---------------------------------------------------------------------------
static __device__ inline v16bf combine16(v8bf lo, v8bf hi) {
  return __builtin_shufflevector(lo, hi, 0, 1, 2, 3, 4, 5, 6, 7,
                                         8, 9, 10, 11, 12, 13, 14, 15);
}

__global__ __launch_bounds__(256)
void gemm_rope(const bf16_t* __restrict__ Xhi, const bf16_t* __restrict__ Xlo,
               const bf16_t* __restrict__ WThi, const bf16_t* __restrict__ WTlo,
               float* __restrict__ out) {
  constexpr int LDA = 40;   // padded bf16 stride: 80B rows -> conflict-free b128 reads
  constexpr int kBufElems = 4 * 128 * LDA;            // Ah|Al|Bh|Bl per buffer
#if USE_ASYNC_LDS
  __shared__ __align__(16) unsigned char smem[2 * kBufElems * sizeof(bf16_t)];
#else
  __shared__ __align__(16) unsigned char smem[kBufElems * sizeof(bf16_t)];
#endif
  float* sC = (float*)smem;            // epilogue reuse: 64*132*4 = 33792 B

  const int tid   = threadIdx.x;
  const int lane  = tid & 31;
  const int wid   = tid >> 5;          // 8 waves
  const int wm    = wid & 3;           // 4 M-strips of 32 rows
  const int wn    = wid >> 2;          // 2 N-strips of 64 cols
  const int lhalf = lane >> 4;
  const int lm    = lane & 15;

  const int n0 = blockIdx.x * 128;     // tile spans exactly one head (128 cols)
  const int m0 = blockIdx.y * 128;

  // staging map: thread -> (row, 16-col half); 32B per thread per buffer
  const int srow = tid >> 1;
  const int scol = (tid & 1) * 16;
  const int sidx = srow * LDA + scol;
  const size_t aBase = (size_t)(m0 + srow) * kD + scol;
  const size_t bBase = (size_t)(n0 + srow) * kD + scol;

  v8f acc[2][4] = {};

#if USE_ASYNC_LDS
  // ---- async double-buffered pipeline: ASYNCcnt-tracked LDS staging ----
  auto stage_async = [&](int buf, int k0) {
    bf16_t* base = (bf16_t*)smem + buf * kBufElems;
    bf16_t* dAh = base;
    bf16_t* dAl = dAh + 128 * LDA;
    bf16_t* dBh = dAl + 128 * LDA;
    bf16_t* dBl = dBh + 128 * LDA;
    GASYNC_B128(Xhi  + aBase + k0,     &dAh[sidx]);
    GASYNC_B128(Xhi  + aBase + k0 + 8, &dAh[sidx + 8]);
    GASYNC_B128(Xlo  + aBase + k0,     &dAl[sidx]);
    GASYNC_B128(Xlo  + aBase + k0 + 8, &dAl[sidx + 8]);
    GASYNC_B128(WThi + bBase + k0,     &dBh[sidx]);
    GASYNC_B128(WThi + bBase + k0 + 8, &dBh[sidx + 8]);
    GASYNC_B128(WTlo + bBase + k0,     &dBl[sidx]);
    GASYNC_B128(WTlo + bBase + k0 + 8, &dBl[sidx + 8]);
  };

  stage_async(0, 0);
  for (int it = 0; it < kD / 32; ++it) {
    const int cur = it & 1;
    if (it + 1 < kD / 32) {
      stage_async(cur ^ 1, (it + 1) * 32);   // prefetch next slab into other buffer
      ASYNC_WAIT(8);                         // in-order: current slab's 8 are done
    } else {
      ASYNC_WAIT(0);
    }
    __syncthreads();                         // all waves' waits done -> slab visible

    bf16_t* base = (bf16_t*)smem + cur * kBufElems;
    const bf16_t* sAh = base;
    const bf16_t* sAl = sAh + 128 * LDA;
    const bf16_t* sBh = sAl + 128 * LDA;
    const bf16_t* sBl = sBh + 128 * LDA;
#else
  // ---- fallback: register staging, single buffer ----
  bf16_t* sAh0 = (bf16_t*)smem;
  bf16_t* sAl0 = sAh0 + 128 * LDA;
  bf16_t* sBh0 = sAl0 + 128 * LDA;
  bf16_t* sBl0 = sBh0 + 128 * LDA;
  for (int k0 = 0; k0 < kD; k0 += 32) {
    __syncthreads();
    {
      const bf16_t* gAh = Xhi  + aBase + k0;
      const bf16_t* gAl = Xlo  + aBase + k0;
      const bf16_t* gBh = WThi + bBase + k0;
      const bf16_t* gBl = WTlo + bBase + k0;
      v8bf a0 = *(const v8bf*)(gAh);     v8bf a1 = *(const v8bf*)(gAh + 8);
      v8bf b0 = *(const v8bf*)(gAl);     v8bf b1 = *(const v8bf*)(gAl + 8);
      v8bf c0 = *(const v8bf*)(gBh);     v8bf c1 = *(const v8bf*)(gBh + 8);
      v8bf d0 = *(const v8bf*)(gBl);     v8bf d1 = *(const v8bf*)(gBl + 8);
      *(v8bf*)&sAh0[sidx] = a0;  *(v8bf*)&sAh0[sidx + 8] = a1;
      *(v8bf*)&sAl0[sidx] = b0;  *(v8bf*)&sAl0[sidx + 8] = b1;
      *(v8bf*)&sBh0[sidx] = c0;  *(v8bf*)&sBh0[sidx + 8] = c1;
      *(v8bf*)&sBl0[sidx] = d0;  *(v8bf*)&sBl0[sidx + 8] = d1;
      if (k0 + 32 < kD) {
        __builtin_prefetch(gAh + 32, 0, 3);
        __builtin_prefetch(gBh + 32, 0, 3);
      }
    }
    __syncthreads();
    const bf16_t* sAh = sAh0;
    const bf16_t* sAl = sAl0;
    const bf16_t* sBh = sBh0;
    const bf16_t* sBl = sBl0;
#endif

    // A fragments: lane = row M, element e -> K = (e/8)*16 + lhalf*8 + e%8
    v16bf ah[2], al[2];
#pragma unroll
    for (int mi = 0; mi < 2; ++mi) {
      const int base_i = (wm * 32 + mi * 16 + lm) * LDA + lhalf * 8;
      ah[mi] = combine16(*(const v8bf*)&sAh[base_i], *(const v8bf*)&sAh[base_i + 16]);
      al[mi] = combine16(*(const v8bf*)&sAl[base_i], *(const v8bf*)&sAl[base_i + 16]);
    }
    // B fragments (WT is K-contiguous per column): lane = col N, same K runs
    v16bf bh[4], bl[4];
#pragma unroll
    for (int ni = 0; ni < 4; ++ni) {
      const int base_i = (wn * 64 + ni * 16 + lm) * LDA + lhalf * 8;
      bh[ni] = combine16(*(const v8bf*)&sBh[base_i], *(const v8bf*)&sBh[base_i + 16]);
      bl[ni] = combine16(*(const v8bf*)&sBl[base_i], *(const v8bf*)&sBl[base_i + 16]);
    }

#pragma unroll
    for (int mi = 0; mi < 2; ++mi)
#pragma unroll
      for (int ni = 0; ni < 4; ++ni) {
        acc[mi][ni] = __builtin_amdgcn_wmma_f32_16x16x32_bf16(
            false, ah[mi], false, bh[ni], (short)0, acc[mi][ni], false, false);
        acc[mi][ni] = __builtin_amdgcn_wmma_f32_16x16x32_bf16(
            false, ah[mi], false, bl[ni], (short)0, acc[mi][ni], false, false);
        acc[mi][ni] = __builtin_amdgcn_wmma_f32_16x16x32_bf16(
            false, al[mi], false, bh[ni], (short)0, acc[mi][ni], false, false);
      }
#if USE_ASYNC_LDS
    __syncthreads();   // all reads of this buffer done before it is async-written
  }
#else
  }
#endif

  // ------------------------- epilogue: RoPE + store -------------------------
  const size_t QSZ = (size_t)kB * kH * kT * kDH;
  int cat;                                  // whole tile is one category
  if (n0 < kNQ) cat = 0; else if (n0 < kNQ + kNK) cat = 1; else cat = 2;

  for (int half = 0; half < 2; ++half) {    // 64-row chunks through LDS
    __syncthreads();
    if ((wm >> 1) == half) {
      const int rbase = (wm & 1) * 32;
#pragma unroll
      for (int mi = 0; mi < 2; ++mi)
#pragma unroll
        for (int ni = 0; ni < 4; ++ni)
#pragma unroll
          for (int i = 0; i < 8; ++i) {
            const int r = rbase + mi * 16 + lhalf * 8 + i;
            const int c = wn * 64 + ni * 16 + lm;
            sC[r * 132 + c] = acc[mi][ni][i];
          }
    }
    __syncthreads();
    for (int idx = tid; idx < 64 * 128; idx += 256) {
      const int r = idx >> 7;
      const int c = idx & 127;               // == head-local d
      const int grow = m0 + half * 64 + r;
      const int b = grow >> 12;              // / kT
      const int t = grow & (kT - 1);
      const float u = sC[r * 132 + c];
      float val = u;
      if (cat != 2) {                        // rotate-half RoPE for Q and K
        const int din = c & 63;
        const float inv = expf(-(float)din * (9.210340372f / 64.0f)); // 10000^(-din/64)
        float sv, cv;
        sincosf((float)t * inv, &sv, &cv);
        const float part = sC[r * 132 + (c ^ 64)];
        val = (c < 64) ? (u * cv - part * sv) : (u * cv + part * sv);
      }
      if (cat == 0) {
        const int h = n0 >> 7;
        out[(((size_t)b * kH + h) * kT + t) * kDH + c] = val;
      } else if (cat == 1) {
        const int hk = (n0 - kNQ) >> 7;
#pragma unroll
        for (int e = 0; e < 4; ++e)          // GQA expansion x4
          out[QSZ + (((size_t)b * kH + hk * 4 + e) * kT + t) * kDH + c] = val;
      } else {
        const int hv = (n0 - kNQ - kNK) >> 7;
#pragma unroll
        for (int e = 0; e < 4; ++e)
          out[2 * QSZ + (((size_t)b * kH + hv * 4 + e) * kT + t) * kDH + c] = val;
      }
    }
  }
}

// ---------------------------------------------------------------------------
extern "C" void kernel_launch(void* const* d_in, const int* in_sizes, int n_in,
                              void* d_out, int out_size, void* d_ws, size_t ws_size,
                              hipStream_t stream) {
  (void)in_sizes; (void)n_in; (void)out_size; (void)ws_size;
  const float* x     = (const float*)d_in[0];
  const float* gamma = (const float*)d_in[1];
  const float* beta  = (const float*)d_in[2];
  const float* Wq    = (const float*)d_in[3];
  const float* Wk    = (const float*)d_in[4];
  const float* Wv    = (const float*)d_in[5];
  float* out = (float*)d_out;

  // workspace layout (bf16): Xhi | Xlo | WThi | WTlo  = 224 MiB total
  bf16_t* Xhi  = (bf16_t*)d_ws;
  bf16_t* Xlo  = Xhi + (size_t)kM * kD;
  bf16_t* WThi = Xlo + (size_t)kM * kD;
  bf16_t* WTlo = WThi + (size_t)kNT * kD;

  ln_to_bf16<<<kM, 256, 0, stream>>>(x, gamma, beta, Xhi, Xlo);

  dim3 gq(kNQ / 32, kD / 32), gk(kNK / 32, kD / 32);
  wT_to_bf16<<<gq, 256, 0, stream>>>(Wq, kNQ, WThi, WTlo);
  wT_to_bf16<<<gk, 256, 0, stream>>>(Wk, kNK, WThi + (size_t)kNQ * kD,
                                               WTlo + (size_t)kNQ * kD);
  wT_to_bf16<<<gk, 256, 0, stream>>>(Wv, kNK, WThi + (size_t)(kNQ + kNK) * kD,
                                               WTlo + (size_t)(kNQ + kNK) * kD);

  dim3 gg(kNT / 128, kM / 128);   // 48 x 64 tiles
  gemm_rope<<<gg, 256, 0, stream>>>(Xhi, Xlo, WThi, WTlo, out);
}
